// GATEncoder_44392781971661
// MI455X (gfx1250) — matrix operand order repaired and checked by reading, not compiled, not run on hardware
//
#include <hip/hip_runtime.h>
#include <hip/hip_bf16.h>
#include <math.h>

// ---------------------------------------------------------------------------
// GAT encoder for MI455X (gfx1250, wave32).
// GEMMs: bf16x3 error-compensated V_WMMA_F32_16X16X32_BF16 (fp32 fidelity),
// with operands PRE-PACKED into WMMA lane layout as bf16 hi/lo arrays so the
// inner loop is pure b128 loads + WMMA (no per-iteration conversion VALU).
// Edge phase: single fused gather/score/scatter pass; softmax denominator
// applied as a post-divide (it is a per-target scalar).
// ---------------------------------------------------------------------------

typedef __attribute__((ext_vector_type(16))) __bf16 v16bf;
typedef __attribute__((ext_vector_type(8)))  float  v8f;

#define V_NODES 100000
#define N_EDGES 1600000
#define NH 4
#define SLOPE 0.2f

// ---------------------------------------------------------------------------
// Pack activation matrix A[M x 256] (f32) into WMMA A-operand order, split
// into bf16 hi + bf16 residual.  A-operand (16-bit A 16x32 per ISA):
//   lane<16: M=lane, K = {k0..k0+7, k0+16..k0+23}
//   lane>=16: M=lane-16, K = {k0+8..k0+15, k0+24..k0+31}
// Stored: ((mtile*8 + ktile)*32 + lane)*16 + i   (i = 0..15, contiguous 32B).
// One block per mtile; wave w = ktile.
// ---------------------------------------------------------------------------
__global__ void __launch_bounds__(256)
pack_a(const float* __restrict__ A, __bf16* __restrict__ hi,
       __bf16* __restrict__ lo, int M) {
  const int kt    = threadIdx.x >> 5;
  const int lane  = threadIdx.x & 31;
  const int mtile = blockIdx.x;
  if (mtile * 16 >= M) return;
  const int lhalf = lane >> 4;
  const int row   = mtile * 16 + (lane & 15);
  const float* ar = A + (size_t)row * 256 + kt * 32 + lhalf * 8;

  float f[16];
#pragma unroll
  for (int i = 0; i < 8; ++i) f[i]     = ar[i];
#pragma unroll
  for (int i = 0; i < 8; ++i) f[8 + i] = ar[16 + i];

  const size_t off = ((size_t)(mtile * 8 + kt) * 32 + lane) * 16;
#pragma unroll
  for (int i = 0; i < 16; ++i) {
    __bf16 h = (__bf16)f[i];
    hi[off + i] = h;
    lo[off + i] = (__bf16)(f[i] - (float)h);
  }
}

// ---------------------------------------------------------------------------
// Pack weight matrix W[256 x 256] (f32, row-major K x N) into WMMA B-operand
// order (16-bit B 32x16): lane<16 -> K {k0..k0+7, k0+16..k0+23}, lane>=16 ->
// K {k0+8..k0+15, k0+24..k0+31}, N = lane&15.  Stored: ((ktile*16 + ntile)*32 + lane)*16 + i.
// 128 waves total -> 16 blocks x 256 threads.
// ---------------------------------------------------------------------------
__global__ void __launch_bounds__(256)
pack_b(const float* __restrict__ W, __bf16* __restrict__ hi,
       __bf16* __restrict__ lo) {
  const int wave  = threadIdx.x >> 5;
  const int lane  = threadIdx.x & 31;
  const int wid   = blockIdx.x * 8 + wave;   // 0..127 = ktile*16 + ntile
  const int ntile = wid & 15;
  const int ktile = wid >> 4;
  const int col   = ntile * 16 + (lane & 15);
  const int kbase = ktile * 32 + (lane >> 4) * 8;
  const size_t off = ((size_t)wid * 32 + lane) * 16;
#pragma unroll
  for (int i = 0; i < 16; ++i) {
    float x = W[(size_t)(kbase + (i < 8 ? i : i + 8)) * 256 + col];
    __bf16 h = (__bf16)x;
    hi[off + i] = h;
    lo[off + i] = (__bf16)(x - (float)h);
  }
}

// ---------------------------------------------------------------------------
// C[M x 256] = act(A @ W + bias) from pre-packed bf16 hi/lo operands.
// grid = (ceil(Mtiles/4), 4), block = 128 (4 waves); wave owns a 16x64 tile.
// Inner loop: 2 A loads + 8 B loads (32B each) + 12 WMMAs per K-step of 32.
// C layout: vgpr i -> M = i + 8*(lane>=16), N = lane&15.
// ---------------------------------------------------------------------------
__global__ void __launch_bounds__(128)
gemm256_wmma_pk(const __bf16* __restrict__ Ahi, const __bf16* __restrict__ Alo,
                const __bf16* __restrict__ Bhi, const __bf16* __restrict__ Blo,
                const float* __restrict__ bias, float* __restrict__ C,
                int M, int act) {
  const int wave  = threadIdx.x >> 5;
  const int lane  = threadIdx.x & 31;
  const int mtile = blockIdx.x * 4 + wave;
  if (mtile * 16 >= M) return;              // wave-uniform: EXEC stays all-1
  const int lrow   = lane & 15;
  const int lhalf  = lane >> 4;
  const int ntbase = blockIdx.y * 4;
  const int n0     = blockIdx.y * 64;

  v8f acc[4] = {};

  for (int kt = 0; kt < 8; ++kt) {
    const size_t aoff = ((size_t)(mtile * 8 + kt) * 32 + lane) * 16;
    const v16bf a_hi = *(const v16bf*)(Ahi + aoff);
    const v16bf a_lo = *(const v16bf*)(Alo + aoff);
#pragma unroll
    for (int nt = 0; nt < 4; ++nt) {
      const size_t boff = ((size_t)(kt * 16 + ntbase + nt) * 32 + lane) * 16;
      const v16bf b_hi = *(const v16bf*)(Bhi + boff);
      const v16bf b_lo = *(const v16bf*)(Blo + boff);
      acc[nt] = __builtin_amdgcn_wmma_f32_16x16x32_bf16(
          false, a_hi, false, b_hi, (short)0, acc[nt], false, false);
      acc[nt] = __builtin_amdgcn_wmma_f32_16x16x32_bf16(
          false, a_hi, false, b_lo, (short)0, acc[nt], false, false);
      acc[nt] = __builtin_amdgcn_wmma_f32_16x16x32_bf16(
          false, a_lo, false, b_hi, (short)0, acc[nt], false, false);
    }
  }

#pragma unroll
  for (int nt = 0; nt < 4; ++nt) {
    const int col = n0 + nt * 16 + lrow;
    const float bv = bias[col];
#pragma unroll
    for (int i = 0; i < 8; ++i) {
      const int r = mtile * 16 + i + lhalf * 8;
      float v = acc[nt][i] + bv;
      if (act) v = v > 0.f ? v : SLOPE * v;
      C[(size_t)r * 256 + col] = v;
    }
  }
}

// ---------------------------------------------------------------------------
// PE projection tables: RP[256][256] = pe_row @ W_in[768:960] (+ b_in),
//                       CP[64][256]  = pe_col @ W_in[960:1152].
// ---------------------------------------------------------------------------
__global__ void pe_proj(const float* __restrict__ W_in,
                        const float* __restrict__ b_in,
                        float* __restrict__ RP, float* __restrict__ CP) {
  __shared__ float pe[192];
  const int r = blockIdx.x;          // 0..319
  const int c = threadIdx.x;         // 0..255
  const bool isRow = r < 256;
  const int pos = isRow ? r : r - 256;
  if (c < 96) {
    float div = expf((float)(2 * c) * (-9.210340371976184f / 192.0f));
    float ang = (float)pos * div;
    pe[2 * c]     = sinf(ang);
    pe[2 * c + 1] = cosf(ang);
  }
  __syncthreads();
  const float* Wb = W_in + (size_t)(isRow ? 768 : 960) * 256 + c;
  float acc = isRow ? b_in[c] : 0.f;
  for (int d = 0; d < 192; ++d) acc += pe[d] * Wb[(size_t)d * 256];
  if (isRow) RP[(size_t)r * 256 + c] = acc;
  else       CP[(size_t)pos * 256 + c] = acc;
}

// ---------------------------------------------------------------------------
// h[v] = relu(LayerNorm(RP[row_idx[v]] + CP[col_idx[v]]))   (b_in folded in RP)
// ---------------------------------------------------------------------------
__global__ void __launch_bounds__(256)
node_init(const int* __restrict__ row_idx, const int* __restrict__ col_idx,
          const float* __restrict__ RP, const float* __restrict__ CP,
          const float* __restrict__ ln_g, const float* __restrict__ ln_b,
          float* __restrict__ h) {
  __shared__ float sbuf[18];
  const int v = blockIdx.x;
  const int c = threadIdx.x;
  const float x = RP[(size_t)row_idx[v] * 256 + c] +
                  CP[(size_t)col_idx[v] * 256 + c];
  float s1 = x, s2 = x * x;
#pragma unroll
  for (int off = 16; off > 0; off >>= 1) {
    s1 += __shfl_xor(s1, off, 32);
    s2 += __shfl_xor(s2, off, 32);
  }
  const int wv = c >> 5;
  if ((c & 31) == 0) { sbuf[wv] = s1; sbuf[8 + wv] = s2; }
  __syncthreads();
  if (c == 0) {
    float a = 0.f, b = 0.f;
#pragma unroll
    for (int i = 0; i < 8; ++i) { a += sbuf[i]; b += sbuf[8 + i]; }
    sbuf[16] = a * (1.f / 256.f);
    sbuf[17] = b * (1.f / 256.f);
  }
  __syncthreads();
  const float mean = sbuf[16];
  const float var  = sbuf[17] - mean * mean;
  float y = (x - mean) * rsqrtf(var + 1e-5f) * ln_g[c] + ln_b[c];
  h[(size_t)v * 256 + c] = y > 0.f ? y : 0.f;
}

// ---------------------------------------------------------------------------
// Fused edge pass: one wave per edge. Lane l owns channels [8l, 8l+8);
// head = l>>3.  a = exp(dot(Q[src,h],K[tgt,h])/8 + leaky(ew*We[h]+be[h]));
// deg[tgt,h] += a;  out[tgt,c] += V[src,c]*a*ew.  (normalize afterwards)
// ---------------------------------------------------------------------------
__global__ void __launch_bounds__(256)
gat_edge(const float* __restrict__ Q, const float* __restrict__ K,
         const float* __restrict__ Vv,
         const int* __restrict__ src, const int* __restrict__ tgt,
         const float* __restrict__ ew,
         const float* __restrict__ We, const float* __restrict__ be,
         float* __restrict__ deg, float* __restrict__ out, int E) {
  const int wid  = (blockIdx.x * blockDim.x + threadIdx.x) >> 5;
  const int lane = threadIdx.x & 31;
  if (wid >= E) return;
  const int s = src[wid];
  const int t = tgt[wid];
  const float w = ew[wid];

  const float* q = Q  + (size_t)s * 256 + lane * 8;
  const float* k = K  + (size_t)t * 256 + lane * 8;
  const float* v = Vv + (size_t)s * 256 + lane * 8;
  __builtin_prefetch(v, 0, 0);   // global_prefetch_b8: V row needed after dot

  float part = 0.f;
#pragma unroll
  for (int i = 0; i < 8; ++i) part += q[i] * k[i];
  part += __shfl_xor(part, 1, 32);
  part += __shfl_xor(part, 2, 32);
  part += __shfl_xor(part, 4, 32);

  const int head = lane >> 3;
  float eb = w * We[head] + be[head];
  eb = eb > 0.f ? eb : SLOPE * eb;
  const float a = __expf(part * 0.125f + eb);   // 1/sqrt(64) = 0.125

  if ((lane & 7) == 0) atomicAdd(&deg[(size_t)t * NH + head], a);

  const float coef = a * w;
  float* o = out + (size_t)t * 256 + lane * 8;
#pragma unroll
  for (int i = 0; i < 8; ++i) atomicAdd(&o[i], v[i] * coef);
}

// out[v][c] /= (deg[v][c/64] + 1e-16)
__global__ void __launch_bounds__(256)
norm_out(float* __restrict__ out, const float* __restrict__ deg) {
  const size_t idx = (size_t)blockIdx.x * 256 + threadIdx.x;
  const int v = (int)(idx >> 8);
  const int c = (int)(idx & 255);
  out[idx] /= (deg[(size_t)v * NH + (c >> 6)] + 1e-16f);
}

// ---------------------------------------------------------------------------
extern "C" void kernel_launch(void* const* d_in, const int* in_sizes, int n_in,
                              void* d_out, int out_size, void* d_ws,
                              size_t ws_size, hipStream_t stream) {
  (void)in_sizes; (void)n_in; (void)out_size; (void)ws_size;
  const int*   row_idx = (const int*)d_in[0];
  const int*   col_idx = (const int*)d_in[1];
  const int*   eidx    = (const int*)d_in[2];
  const float* eweight = (const float*)d_in[3];
  const float* W_in    = (const float*)d_in[4];
  const float* b_in    = (const float*)d_in[5];
  const float* ln_g    = (const float*)d_in[6];
  const float* ln_b    = (const float*)d_in[7];
  const float* Wq = (const float*)d_in[8];   const float* bq = (const float*)d_in[9];
  const float* Wk = (const float*)d_in[10];  const float* bk = (const float*)d_in[11];
  const float* Wv = (const float*)d_in[12];  const float* bv = (const float*)d_in[13];
  const float* Wo = (const float*)d_in[14];  const float* bo = (const float*)d_in[15];
  const float* We = (const float*)d_in[16];  const float* be = (const float*)d_in[17];

  const int V = V_NODES, E = N_EDGES;
  const size_t NS = (size_t)V * 256;

  float* ws = (float*)d_ws;
  float* B0  = ws;                      // 4 rotating V x 256 f32 buffers
  float* B1  = B0 + NS;
  float* B2  = B1 + NS;
  float* B3  = B2 + NS;
  float* APKf = B3 + NS;                // NS floats = 2*NS bf16 (hi + lo)
  float* WPKf = APKf + NS;              // 65536 floats = 2*65536 bf16
  float* deg = WPKf + 65536;            // V x 4
  float* RP  = deg + (size_t)V * NH;    // 256 x 256
  float* CP  = RP + 256 * 256;          // 64 x 256

  __bf16* Ahi = (__bf16*)APKf;          __bf16* Alo = Ahi + NS;
  __bf16* Bhi = (__bf16*)WPKf;          __bf16* Blo = Bhi + 65536;

  const int* srcp = eidx;
  const int* tgtp = eidx + E;

  // --- input projection + LayerNorm + ReLU ---
  pe_proj<<<320, 256, 0, stream>>>(W_in, b_in, RP, CP);
  node_init<<<V, 256, 0, stream>>>(row_idx, col_idx, RP, CP, ln_g, ln_b, B0);

  const dim3 gg((6250 + 3) / 4, 4);     // 6250 M-tiles, 4 waves/block; 4 N-blocks
  float* hin = B0;
  for (int l = 0; l < 2; ++l) {
    float* qb = (hin == B0) ? B1 : B0;
    float* kb = B2;
    float* vb = B3;

    // pack activations once, then Q/K/V GEMMs share the packed A
    pack_a<<<6250, 256, 0, stream>>>(hin, Ahi, Alo, V);
    pack_b<<<16, 256, 0, stream>>>(Wq + l * 65536, Bhi, Blo);
    gemm256_wmma_pk<<<gg, 128, 0, stream>>>(Ahi, Alo, Bhi, Blo, bq + l * 256, qb, V, 0);
    pack_b<<<16, 256, 0, stream>>>(Wk + l * 65536, Bhi, Blo);
    gemm256_wmma_pk<<<gg, 128, 0, stream>>>(Ahi, Alo, Bhi, Blo, bk + l * 256, kb, V, 0);
    pack_b<<<16, 256, 0, stream>>>(Wv + l * 65536, Bhi, Blo);
    gemm256_wmma_pk<<<gg, 128, 0, stream>>>(Ahi, Alo, Bhi, Blo, bv + l * 256, vb, V, 0);

    // hin (f32) is dead after pack_a -> reuse it as the message accumulator
    hipMemsetAsync(deg, 0, (size_t)V * NH * sizeof(float), stream);
    hipMemsetAsync(hin, 0, NS * sizeof(float), stream);

    gat_edge<<<E / 8, 256, 0, stream>>>(qb, kb, vb, srcp, tgtp, eweight,
                                        We + l * NH, be + l * NH, deg, hin, E);
    norm_out<<<V, 256, 0, stream>>>(hin, deg);

    pack_a<<<6250, 256, 0, stream>>>(hin, Ahi, Alo, V);
    pack_b<<<16, 256, 0, stream>>>(Wo + l * 65536, Bhi, Blo);
    float* dst = (l == 1) ? (float*)d_out : qb;
    gemm256_wmma_pk<<<gg, 128, 0, stream>>>(Ahi, Alo, Bhi, Blo, bo + l * 256, dst, V, 1);
    hin = qb;
  }
}
